// AttentiveBP_83322365542496
// MI455X (gfx1250) — matrix operand order repaired and checked by reading, not compiled, not run on hardware
//
#include <hip/hip_runtime.h>
#include <math.h>

// ---------------------------------------------------------------------------
// CDNA5 (gfx1250) WMMA types
// ---------------------------------------------------------------------------
typedef __attribute__((ext_vector_type(16))) _Float16 v16h;
typedef __attribute__((ext_vector_type(8)))  float    v8f;

#define CDIV(a, b) (((a) + (b) - 1) / (b))

__device__ __forceinline__ float lrelu(float v, float s) { return v > 0.f ? v : s * v; }
__device__ __forceinline__ float sigmoidf(float v) { return 1.f / (1.f + expf(-v)); }

// float atomic max via signed/unsigned integer compare trick
__device__ __forceinline__ void atomicMaxFloat(float* addr, float v) {
  if (v >= 0.f) atomicMax((int*)addr, __float_as_int(v));
  else          atomicMin((unsigned int*)addr, __float_as_uint(v));
}

__device__ __forceinline__ void pack4(v16h& v, int o, float4 f) {
  v[o + 0] = (_Float16)f.x;
  v[o + 1] = (_Float16)f.y;
  v[o + 2] = (_Float16)f.z;
  v[o + 3] = (_Float16)f.w;
}

// ---------------------------------------------------------------------------
// WMMA GEMM:  C[M,N] = A[M,K] * Bt^T, with Bt stored [N,K] row-major.
// (For X@W with W[K,N], pre-transpose W once; GRU weights already [N,K].)
// Requirements (all satisfied here): M%16==0, N%32==0, K%32==0.
// One wave per 16x32 output tile (2 col-tiles share the A fragment).
// Fragment layouts used here (contiguous k, same for A and B):
//   A lane(l): row bm*16+(l&15); elems e -> K=kk+16*(l>>4)+e (contig 16)
//              => 4x float4
//   B lane(l): col (l&15); elems e -> K=kk+16*(l>>4)+e (contig 16) => 4x float4
// All loads are 16B-aligned since K%32==0 and bases are 256B-aligned.
// ---------------------------------------------------------------------------
__global__ __launch_bounds__(32)
void gemm_wmma(const float* __restrict__ A, const float* __restrict__ Bt,
               float* __restrict__ C, int M_, int N_, int K_) {
  const int bn   = blockIdx.x * 2;      // first of two 16-col tiles
  const int bm   = blockIdx.y;          // 16-row tile
  const int lane = threadIdx.x & 31;
  const int hsel = lane >> 4;           // 0: lanes 0-15, 1: lanes 16-31
  const int l15  = lane & 15;

  const float* __restrict__ Arow  = A  + (size_t)(bm * 16 + l15) * K_;
  const float* __restrict__ Brow0 = Bt + (size_t)(bn * 16 + l15) * K_;
  const float* __restrict__ Brow1 = Bt + (size_t)((bn + 1) * 16 + l15) * K_;

  v8f acc0 = {}, acc1 = {};
  for (int kk = 0; kk < K_; kk += 32) {
    if (kk + 32 < K_) {                 // speculative prefetch of next k-slab
      __builtin_prefetch(Arow + kk + 32, 0, 1);
      __builtin_prefetch(Brow0 + kk + 32, 0, 1);
      __builtin_prefetch(Brow1 + kk + 32, 0, 1);
    }
    // ---- A fragment: 4x float4 (one contiguous 16-float run) ----
    const float4* apA = (const float4*)(Arow + kk + (hsel << 4));
    const float4* apB = apA + 2;
    float4 a0 = apA[0], a1 = apA[1], a2 = apB[0], a3 = apB[1];
    // ---- B fragments: 4x float4 each (one contiguous 16-float run) ----
    const float4* bp0 = (const float4*)(Brow0 + kk + (hsel << 4));
    const float4* bp1 = (const float4*)(Brow1 + kk + (hsel << 4));
    float4 c0 = bp0[0], c1 = bp0[1], c2 = bp0[2], c3 = bp0[3];
    float4 d0 = bp1[0], d1 = bp1[1], d2 = bp1[2], d3 = bp1[3];

    v16h a, b0, b1;
    pack4(a, 0, a0);  pack4(a, 4, a1);  pack4(a, 8, a2);  pack4(a, 12, a3);
    pack4(b0, 0, c0); pack4(b0, 4, c1); pack4(b0, 8, c2); pack4(b0, 12, c3);
    pack4(b1, 0, d0); pack4(b1, 4, d1); pack4(b1, 8, d2); pack4(b1, 12, d3);

    acc0 = __builtin_amdgcn_wmma_f32_16x16x32_f16(false, a, false, b0,
                                                  (short)0, acc0, false, false);
    acc1 = __builtin_amdgcn_wmma_f32_16x16x32_f16(false, a, false, b1,
                                                  (short)0, acc1, false, false);
  }
  // D layout: lane -> col (l15); VGPR r -> row r + 8*hsel
#pragma unroll
  for (int r = 0; r < 8; ++r) {
    size_t row = (size_t)(bm * 16 + r + (hsel << 3)) * N_;
    C[row + bn * 16 + l15]      = acc0[r];
    C[row + (bn + 1) * 16 + l15] = acc1[r];
  }
}

// small weight transpose: src [K,N] -> dst [N,K]
__global__ void transpose_small(const float* __restrict__ src, float* __restrict__ dst,
                                int K, int N) {
  int i = blockIdx.x * blockDim.x + threadIdx.x;
  if (i >= K * N) return;
  int k = i / N, n = i % N;
  dst[n * K + k] = src[i];
}

// ---------------------------------------------------------------------------
// Elementwise / scatter kernels
// ---------------------------------------------------------------------------
__global__ void fillf(float* p, float v, int n) {
  int i = blockIdx.x * blockDim.x + threadIdx.x;
  if (i < n) p[i] = v;
}

// GRUCell gates: hout = (1-z)*n + z*hprev
__global__ void gru_gate(const float* __restrict__ gi, const float* __restrict__ gh,
                         const float* __restrict__ bih, const float* __restrict__ bhh,
                         const float* __restrict__ hprev, float* __restrict__ hout,
                         int rows, int hs) {
  int i = blockIdx.x * blockDim.x + threadIdx.x;
  if (i >= rows * hs) return;
  int r = i / hs, j = i % hs;
  size_t base = (size_t)r * 3 * hs;
  float ir = gi[base + j]          + bih[j];
  float iz = gi[base + hs + j]     + bih[hs + j];
  float in_= gi[base + 2 * hs + j] + bih[2 * hs + j];
  float hr = gh[base + j]          + bhh[j];
  float hz = gh[base + hs + j]     + bhh[hs + j];
  float hn = gh[base + 2 * hs + j] + bhh[2 * hs + j];
  float rg = sigmoidf(ir + hr);
  float z  = sigmoidf(iz + hz);
  float n  = tanhf(in_ + rg * hn);
  hout[(size_t)r * hs + j] = (1.f - z) * n + z * hprev[(size_t)r * hs + j];
}

// edge_attr rows [0,E1): [prefix(32) | costs(32) | hidden1(64)]
__global__ void build_ea1(const float* __restrict__ pre, const float* __restrict__ cst,
                          const float* __restrict__ h1, float* __restrict__ ea, int E1) {
  int i = blockIdx.x * blockDim.x + threadIdx.x;
  if (i >= E1 * 128) return;
  int e = i / 128, j = i % 128;
  float v;
  if (j < 32)       v = pre[(size_t)e * 32 + j];
  else if (j < 64)  v = cst[(size_t)e * 32 + (j - 32)];
  else              v = h1[(size_t)e * 64 + (j - 64)];
  ea[i] = v;
}

// edge_attr rows [E1,E): hidden2
__global__ void build_ea2(const float* __restrict__ h2, float* __restrict__ ea,
                          int E1, int E2) {
  int i = blockIdx.x * blockDim.x + threadIdx.x;
  if (i >= E2 * 128) return;
  ea[(size_t)E1 * 128 + i] = h2[i];
}

__global__ void count_in_edges(const int* __restrict__ ei, float* __restrict__ cnt, int E) {
  int e = blockIdx.x * blockDim.x + threadIdx.x;
  if (e < E) atomicAdd(&cnt[ei[E + e]], 1.f);
}

// self-loop fill_value='mean': loop_attr[n] = mean of incoming edge attrs
__global__ void loop_attr_sum(const float* __restrict__ ea_in, const int* __restrict__ ei,
                              float* __restrict__ ea, int E) {
  int i = blockIdx.x * blockDim.x + threadIdx.x;
  if (i >= E * 128) return;
  int e = i / 128, j = i % 128;
  int d = ei[E + e];
  atomicAdd(&ea[((size_t)E + d) * 128 + j], ea_in[(size_t)e * 128 + j]);
}

__global__ void loop_attr_div(float* __restrict__ ea, const float* __restrict__ cnt,
                              int E, int N) {
  int i = blockIdx.x * blockDim.x + threadIdx.x;
  if (i >= N * 128) return;
  int n = i / 128;
  ea[((size_t)E + n) * 128 + (i % 128)] /= fmaxf(cnt[n], 1.f);
}

// per-node attention scores: as_n[n,h] = sum_c h[n,h*C+c]*a_s[h,c]
__global__ void node_scores(const float* __restrict__ hl, const float* __restrict__ a_s,
                            const float* __restrict__ a_d, float* __restrict__ as_n,
                            float* __restrict__ ad_n, int N, int Cc) {
  int i = blockIdx.x * blockDim.x + threadIdx.x;
  if (i >= N * 4) return;
  int n = i / 4, h = i % 4;
  float s = 0.f, d = 0.f;
  for (int c = 0; c < Cc; ++c) {
    float hv = hl[(size_t)n * 4 * Cc + h * Cc + c];
    s += hv * a_s[h * Cc + c];
    d += hv * a_d[h * Cc + c];
  }
  as_n[i] = s; ad_n[i] = d;
}

// alpha = leaky_relu(as[src]+ad[dst]+e.ae, 0.2); also segment max over dst
__global__ void alpha_raw(const float* __restrict__ as_n, const float* __restrict__ ad_n,
                          const float* __restrict__ el, const float* __restrict__ ae,
                          const int* __restrict__ ei, int E, int Etot, int Cc,
                          float* __restrict__ alpha, float* __restrict__ segmax) {
  int i = blockIdx.x * blockDim.x + threadIdx.x;
  if (i >= Etot * 4) return;
  int e = i / 4, h = i % 4;
  int s, d;
  if (e < E) { s = ei[e]; d = ei[E + e]; } else { s = d = e - E; }
  float ev = 0.f;
  for (int c = 0; c < Cc; ++c)
    ev += el[(size_t)e * 4 * Cc + h * Cc + c] * ae[h * Cc + c];
  float v = lrelu(as_n[s * 4 + h] + ad_n[d * 4 + h] + ev, 0.2f);
  alpha[i] = v;
  atomicMaxFloat(&segmax[d * 4 + h], v);
}

__global__ void alpha_exp(float* __restrict__ alpha, const float* __restrict__ segmax,
                          float* __restrict__ segsum, const int* __restrict__ ei,
                          int E, int Etot) {
  int i = blockIdx.x * blockDim.x + threadIdx.x;
  if (i >= Etot * 4) return;
  int e = i / 4, h = i % 4;
  int d = (e < E) ? ei[E + e] : (e - E);
  float ex = expf(alpha[i] - segmax[d * 4 + h]);
  alpha[i] = ex;
  atomicAdd(&segsum[d * 4 + h], ex);
}

// out[dst] += h[src] * alpha/(segsum+eps)
__global__ void gat_aggregate(const float* __restrict__ hl, const float* __restrict__ alpha,
                              const float* __restrict__ segsum, const int* __restrict__ ei,
                              int E, int Etot, int Cc, float* __restrict__ out) {
  int i = blockIdx.x * blockDim.x + threadIdx.x;
  int HC = 4 * Cc;
  if (i >= Etot * HC) return;
  int e = i / HC, j = i % HC, h = j / Cc;
  int s, d;
  if (e < E) { s = ei[e]; d = ei[E + e]; } else { s = d = e - E; }
  float w = alpha[e * 4 + h] / (segsum[d * 4 + h] + 1e-16f);
  atomicAdd(&out[(size_t)d * HC + j], hl[(size_t)s * HC + j] * w);
}

__global__ void gat_finalize_concat(const float* __restrict__ agg, const float* __restrict__ b,
                                    float* __restrict__ out, int N, int HC) {
  int i = blockIdx.x * blockDim.x + threadIdx.x;
  if (i >= N * HC) return;
  out[i] = lrelu(agg[i] + b[i % HC], 0.01f);
}

__global__ void gat_finalize_mean(const float* __restrict__ agg, const float* __restrict__ b,
                                  float* __restrict__ out, int N) {
  int i = blockIdx.x * blockDim.x + threadIdx.x;
  if (i >= N * 32) return;
  int n = i / 32, c = i % 32;
  size_t base = (size_t)n * 128;
  float v = 0.25f * (agg[base + c] + agg[base + 32 + c] +
                     agg[base + 64 + c] + agg[base + 96 + c]) + b[c];
  out[i] = lrelu(v, 0.01f);
}

__global__ void pool_scatter(const float* __restrict__ h4, const int* __restrict__ sidx,
                             float* __restrict__ pool, int N) {
  int i = blockIdx.x * blockDim.x + threadIdx.x;
  if (i >= N * 32) return;
  int n = i / 32, c = i % 32;
  atomicAdd(&pool[(size_t)sidx[n] * 32 + c], h4[i]);
}

__global__ void pool_finalize(const float* __restrict__ pool, const float* __restrict__ dom,
                              float* __restrict__ poolf, int M) {
  int i = blockIdx.x * blockDim.x + threadIdx.x;
  if (i >= M * 32) return;
  int m = i / 32, c = i % 32;
  poolf[i] = pool[(size_t)(m + 1) * 32 + c] / dom[m];
}

// wv[h,d] = sum_e Wx[h,d,e] * Ws[h, off+e]   (folds Wq/Wk with scoring vec)
__global__ void proj_fold(const float* __restrict__ Wx, const float* __restrict__ Ws,
                          int off, float* __restrict__ wv) {
  int i = blockIdx.x * blockDim.x + threadIdx.x;
  if (i >= 4 * 32) return;
  int h = i / 32, d = i % 32;
  float s = 0.f;
  for (int e = 0; e < 32; ++e)
    s += Wx[(size_t)h * 32 * 32 + d * 32 + e] * Ws[h * 64 + off + e];
  wv[i] = s;
}

// qs[m,h] = pool[m,:] . wv[h,:]
__global__ void score_rows(const float* __restrict__ poolf, const float* __restrict__ wv,
                           float* __restrict__ qs, int M) {
  int i = blockIdx.x * blockDim.x + threadIdx.x;
  if (i >= M * 4) return;
  int m = i / 4, h = i % 4;
  float s = 0.f;
  for (int d = 0; d < 32; ++d) s += poolf[m * 32 + d] * wv[h * 32 + d];
  qs[i] = s;
}

// per-group softmax over K neighbors, scaled by (n_targets-1)
__global__ void group_softmax(const float* __restrict__ qs, const float* __restrict__ ks,
                              const float* __restrict__ bs, const int* __restrict__ nsrc,
                              const int* __restrict__ nidx, const int* __restrict__ ntgt,
                              float* __restrict__ outw, int M, int Kn) {
  __shared__ float sv[256];
  __shared__ float mx[4], sm[4];
  int g = blockIdx.x;
  int t = threadIdx.x;                    // t = k*4 + h
  if (t >= Kn * 4) return;
  int k = t / 4, h = t % 4;
  int i = nidx[g];
  int j = nsrc[(size_t)g * Kn + k];
  float s = sigmoidf(qs[i * 4 + h] + ks[j * 4 + h] + bs[h]);
  sv[t] = s;
  __syncthreads();
  if (t < 4) {
    float m = -INFINITY;
    for (int kk = 0; kk < Kn; ++kk) m = fmaxf(m, sv[kk * 4 + t]);
    float su = 0.f;
    for (int kk = 0; kk < Kn; ++kk) su += expf(sv[kk * 4 + t] - m);
    mx[t] = m; sm[t] = su;
  }
  __syncthreads();
  float fac = (float)(ntgt[0] - 1);
  outw[(size_t)g * Kn * 4 + k * 4 + h] = expf(sv[t] - mx[h]) / sm[h] * fac;
}

// ---------------------------------------------------------------------------
// Orchestration
// ---------------------------------------------------------------------------
static inline void launch_gemm(const float* A, const float* Bt, float* C,
                               int M_, int N_, int K_, hipStream_t s) {
  dim3 grid(N_ / 32, M_ / 16);
  gemm_wmma<<<grid, 32, 0, s>>>(A, Bt, C, M_, N_, K_);
}
static inline void launch_fill(float* p, float v, long long n, hipStream_t s) {
  fillf<<<CDIV((int)n, 256), 256, 0, s>>>(p, v, (int)n);
}

extern "C" void kernel_launch(void* const* d_in, const int* in_sizes, int n_in,
                              void* d_out, int out_size, void* d_ws, size_t ws_size,
                              hipStream_t stream) {
  // ---- inputs (setup_inputs order) ----
  const float* x        = (const float*)d_in[0];
  const int*   ei       = (const int*)d_in[1];
  const float* prefix   = (const float*)d_in[2];
  const float* costs    = (const float*)d_in[3];
  const float* msg1     = (const float*)d_in[4];
  const float* hid1_in  = (const float*)d_in[5];
  const float* msg2     = (const float*)d_in[6];
  const float* hid2_in  = (const float*)d_in[7];
  const int*   sidx     = (const int*)d_in[8];
  const float* dom      = (const float*)d_in[9];
  const int*   nsrc     = (const int*)d_in[10];
  const int*   nidx     = (const int*)d_in[11];
  const int*   ntgt     = (const int*)d_in[12];
  const float* W_ih1 = (const float*)d_in[13];
  const float* W_hh1 = (const float*)d_in[14];
  const float* b_ih1 = (const float*)d_in[15];
  const float* b_hh1 = (const float*)d_in[16];
  const float* W_ih2 = (const float*)d_in[17];
  const float* W_hh2 = (const float*)d_in[18];
  const float* b_ih2 = (const float*)d_in[19];
  const float* b_hh2 = (const float*)d_in[20];
  const float* Wl[4]  = {(const float*)d_in[21], (const float*)d_in[27],
                         (const float*)d_in[33], (const float*)d_in[39]};
  const float* asl[4] = {(const float*)d_in[22], (const float*)d_in[28],
                         (const float*)d_in[34], (const float*)d_in[40]};
  const float* adl[4] = {(const float*)d_in[23], (const float*)d_in[29],
                         (const float*)d_in[35], (const float*)d_in[41]};
  const float* Wel[4] = {(const float*)d_in[24], (const float*)d_in[30],
                         (const float*)d_in[36], (const float*)d_in[42]};
  const float* ael[4] = {(const float*)d_in[25], (const float*)d_in[31],
                         (const float*)d_in[37], (const float*)d_in[43]};
  const float* bl[4]  = {(const float*)d_in[26], (const float*)d_in[32],
                         (const float*)d_in[38], (const float*)d_in[44]};
  const float* Wq = (const float*)d_in[45];
  const float* Wk = (const float*)d_in[46];
  const float* Ws = (const float*)d_in[47];
  const float* bs = (const float*)d_in[48];

  // ---- dims ----
  const int N    = in_sizes[0] / 64;       // 100000
  const int E    = in_sizes[1] / 2;        // 600000
  const int E1   = in_sizes[2] / 32;       // 300000
  const int E2   = in_sizes[7] / 128;      // 300000
  const int Etot = E + N;                  // 700000
  const int M    = in_sizes[9];            // 512
  const int G    = in_sizes[11];           // 4096
  const int Kn   = in_sizes[10] / G;       // 16

  // ---- outputs: weights [G,Kn,4] | hidden1 [E1,64] | hidden2 [E2,128] ----
  float* outw = (float*)d_out;
  float* hidden1 = outw + (size_t)G * Kn * 4;
  float* hidden2 = hidden1 + (size_t)E1 * 64;

  // ---- workspace (bump-allocated, phases reuse; peak = GRU: 2*E2*384 fl) ----
  float* ws = (float*)d_ws;
  float* gi = ws;
  float* gh = ws + (size_t)E2 * 384;
  float* tailw = gh + (size_t)E2 * 384;              // small persistent tail
  float* wT  = tailw;                                // <= 128*128
  float* weT = wT + 128 * 128;                       // <= 128*128

  // ===== Phase 1: GRU cells (WMMA GEMMs + gate fusion) =====
  // GRU weights are [3H, K] row-major == exactly the Bt layout we need.
  launch_gemm(msg1,    W_ih1, gi, E1, 192, 64,  stream);
  launch_gemm(hid1_in, W_hh1, gh, E1, 192, 64,  stream);
  gru_gate<<<CDIV(E1 * 64, 256), 256, 0, stream>>>(gi, gh, b_ih1, b_hh1,
                                                   hid1_in, hidden1, E1, 64);
  launch_gemm(msg2,    W_ih2, gi, E2, 384, 64,  stream);
  launch_gemm(hid2_in, W_hh2, gh, E2, 384, 128, stream);
  gru_gate<<<CDIV(E2 * 128, 256), 256, 0, stream>>>(gi, gh, b_ih2, b_hh2,
                                                    hid2_in, hidden2, E2, 128);

  // ===== Phase 2: edge_attr [Etot,128] with self-loop mean fill =====
  float* ea     = ws;                                // Etot*128
  float* elin   = ea     + (size_t)Etot * 128;       // Etot*128 (layer-4 max)
  float* nbufA  = elin   + (size_t)Etot * 128;       // N*128
  float* nbufB  = nbufA  + (size_t)N * 128;          // N*128
  float* as_n   = nbufB  + (size_t)N * 128;          // N*4
  float* ad_n   = as_n   + (size_t)N * 4;
  float* alpha  = ad_n   + (size_t)N * 4;            // Etot*4
  float* segmax = alpha  + (size_t)Etot * 4;         // N*4
  float* segsum = segmax + (size_t)N * 4;            // N*4
  float* cnt    = segsum + (size_t)N * 4;            // N
  float* pool   = cnt    + (size_t)N;                // (M+1)*32
  float* poolf  = pool   + (size_t)(M + 1) * 32;     // M*32
  float* wqv    = poolf  + (size_t)M * 32;           // 128
  float* wkv    = wqv + 128;                         // 128
  float* qsb    = wkv + 128;                         // M*4
  float* ksb    = qsb + (size_t)M * 4;               // M*4

  build_ea1<<<CDIV(E1 * 128, 256), 256, 0, stream>>>(prefix, costs, hidden1, ea, E1);
  build_ea2<<<CDIV(E2 * 128, 256), 256, 0, stream>>>(hidden2, ea, E1, E2);
  launch_fill(ea + (size_t)E * 128, 0.f, (long long)N * 128, stream);
  launch_fill(cnt, 0.f, N, stream);
  count_in_edges<<<CDIV(E, 256), 256, 0, stream>>>(ei, cnt, E);
  loop_attr_sum<<<CDIV(E * 128, 256), 256, 0, stream>>>(ea, ei, ea, E);
  loop_attr_div<<<CDIV(N * 128, 256), 256, 0, stream>>>(ea, cnt, E, N);

  // ===== Phase 3: 4 GAT layers =====
  const int Kin[4] = {64, 32, 32, 32};
  const int Cc[4]  = {8, 8, 8, 32};                  // per-head channels
  const float* xin = x;
  for (int l = 0; l < 4; ++l) {
    int HC  = 4 * Cc[l];                             // node out width
    int HCe = HC;                                    // edge out width
    float* hl  = nbufA;                              // node transform
    float* agg = nbufB;                              // aggregation target
    // pre-transpose tiny weights -> Bt layout for the vectorized GEMM
    transpose_small<<<CDIV(Kin[l] * HC, 256), 256, 0, stream>>>(Wl[l], wT, Kin[l], HC);
    transpose_small<<<CDIV(128 * HCe, 256), 256, 0, stream>>>(Wel[l], weT, 128, HCe);
    launch_gemm(xin, wT,  hl,   N,    HC,  Kin[l], stream);
    launch_gemm(ea,  weT, elin, Etot, HCe, 128,    stream);
    node_scores<<<CDIV(N * 4, 256), 256, 0, stream>>>(hl, asl[l], adl[l],
                                                      as_n, ad_n, N, Cc[l]);
    launch_fill(segmax, -INFINITY, (long long)N * 4, stream);
    launch_fill(segsum, 0.f, (long long)N * 4, stream);
    alpha_raw<<<CDIV(Etot * 4, 256), 256, 0, stream>>>(as_n, ad_n, elin, ael[l],
                                                       ei, E, Etot, Cc[l],
                                                       alpha, segmax);
    alpha_exp<<<CDIV(Etot * 4, 256), 256, 0, stream>>>(alpha, segmax, segsum,
                                                       ei, E, Etot);
    launch_fill(agg, 0.f, (long long)N * HC, stream);
    gat_aggregate<<<CDIV(Etot * HC, 256), 256, 0, stream>>>(hl, alpha, segsum,
                                                            ei, E, Etot, Cc[l], agg);
    if (l < 3) {
      gat_finalize_concat<<<CDIV(N * HC, 256), 256, 0, stream>>>(agg, bl[l], hl, N, HC);
    } else {
      gat_finalize_mean<<<CDIV(N * 32, 256), 256, 0, stream>>>(agg, bl[l], hl, N);
    }
    xin = hl;                                        // next layer input (in nbufA)
  }

  // ===== Phase 4: segment pooling to [M,32] =====
  launch_fill(pool, 0.f, (long long)(M + 1) * 32, stream);
  pool_scatter<<<CDIV(N * 32, 256), 256, 0, stream>>>(nbufA, sidx, pool, N);
  pool_finalize<<<CDIV(M * 32, 256), 256, 0, stream>>>(pool, dom, poolf, M);

  // ===== Phase 5: factorized pairwise attention scores =====
  proj_fold<<<1, 128, 0, stream>>>(Wq, Ws, 0,  wqv);
  proj_fold<<<1, 128, 0, stream>>>(Wk, Ws, 32, wkv);
  score_rows<<<CDIV(M * 4, 256), 256, 0, stream>>>(poolf, wqv, qsb, M);
  score_rows<<<CDIV(M * 4, 256), 256, 0, stream>>>(poolf, wkv, ksb, M);

  // ===== Phase 6: per-group softmax over neighbor sets =====
  group_softmax<<<G, Kn * 4, 0, stream>>>(qsb, ksb, bs, nsrc, nidx, ntgt,
                                          outw, M, Kn);
}
